// BaselineT_13486197310027
// MI455X (gfx1250) — matrix operand order repaired and checked by reading, not compiled, not run on hardware
//
#include <hip/hip_runtime.h>
#include <hip/hip_bf16.h>

#define T_ 2048
#define D_ 1024
#define H_ 16
#define HD_ 64
#define DF_ 4096
#define V_ 32000
#define L_ 4

typedef __attribute__((ext_vector_type(16))) __bf16 v16bf;
typedef __attribute__((ext_vector_type(8)))  __bf16 bf16x8;
typedef __attribute__((ext_vector_type(4)))  __bf16 bf16x4;
typedef __attribute__((ext_vector_type(2)))  __bf16 bf16x2;
typedef __attribute__((ext_vector_type(8)))  float  v8f;

union ABFrag { v16bf v; bf16x8 h[2]; };

__device__ __forceinline__ __bf16 tobf(float f) {
    unsigned u = __builtin_bit_cast(unsigned, f);
    u += 0x7FFFu + ((u >> 16) & 1u);           // round-to-nearest-even
    unsigned short s = (unsigned short)(u >> 16);
    return __builtin_bit_cast(__bf16, s);
}

// packed f32x2 -> bf16x2 (hardware v_cvt_pk_bf16_f32 when available)
__device__ __forceinline__ bf16x2 cvt2(float a, float b) {
#if __has_builtin(__builtin_amdgcn_cvt_pk_bf16_f32)
    return __builtin_amdgcn_cvt_pk_bf16_f32(a, b);
#else
    bf16x2 r; r[0] = tobf(a); r[1] = tobf(b); return r;
#endif
}

__device__ __forceinline__ bf16x4 tobf4(float4 f) {
    union { bf16x4 v; bf16x2 h[2]; } u;
    u.h[0] = cvt2(f.x, f.y);
    u.h[1] = cvt2(f.z, f.w);
    return u.v;
}

__device__ __forceinline__ bf16x8 load8bf(const float* p) {
    float4 a = *(const float4*)p;
    float4 b = *(const float4*)(p + 4);
    union { bf16x8 v; bf16x2 h[4]; } u;
    u.h[0] = cvt2(a.x, a.y);
    u.h[1] = cvt2(a.z, a.w);
    u.h[2] = cvt2(b.x, b.y);
    u.h[3] = cvt2(b.z, b.w);
    return u.v;
}

// ---------------------------------------------------------------- embedding
__global__ __launch_bounds__(256) void embed_kernel(
    const int* __restrict__ ids, const float* __restrict__ te,
    const float* __restrict__ pe, float* __restrict__ x)
{
    int t = blockIdx.x;
    int c = threadIdx.x * 4;
    int id = ids[t];
    float4 a = *(const float4*)(te + (size_t)id * D_ + c);
    float4 p = *(const float4*)(pe + (size_t)t  * D_ + c);
    a.x += p.x; a.y += p.y; a.z += p.z; a.w += p.w;
    *(float4*)(x + (size_t)t * D_ + c) = a;
}

// ---------------------------------------------------------------- layernorm
__global__ __launch_bounds__(256) void ln_kernel(
    const float* __restrict__ x, const float* __restrict__ w,
    const float* __restrict__ b, float* __restrict__ y)
{
    __shared__ float red[16];
    int row = blockIdx.x;
    int c = threadIdx.x * 4;
    const float* xp = x + (size_t)row * D_;
    float4 v = *(const float4*)(xp + c);
    float s  = v.x + v.y + v.z + v.w;
    float s2 = v.x*v.x + v.y*v.y + v.z*v.z + v.w*v.w;
    #pragma unroll
    for (int d = 1; d < 32; d <<= 1) {
        s  += __shfl_xor(s,  d, 32);
        s2 += __shfl_xor(s2, d, 32);
    }
    int wid = threadIdx.x >> 5, lane = threadIdx.x & 31;
    if (lane == 0) { red[wid] = s; red[8 + wid] = s2; }
    __syncthreads();
    float ts = 0.f, ts2 = 0.f;
    #pragma unroll
    for (int i = 0; i < 8; ++i) { ts += red[i]; ts2 += red[8 + i]; }
    float mean = ts * (1.0f / D_);
    float var  = ts2 * (1.0f / D_) - mean * mean;
    float inv  = rsqrtf(var + 1e-5f);
    float4 wv = *(const float4*)(w + c);
    float4 bv = *(const float4*)(b + c);
    float4 o;
    o.x = (v.x - mean) * inv * wv.x + bv.x;
    o.y = (v.y - mean) * inv * wv.y + bv.y;
    o.z = (v.z - mean) * inv * wv.z + bv.z;
    o.w = (v.w - mean) * inv * wv.w + bv.w;
    *(float4*)(y + (size_t)row * D_ + c) = o;
}

// ---------------------------------------------------------------- GEMM
// C[M,N] = act(A[M,K] * W + bias) (+ residual).  fp32 in HBM, bf16 in LDS,
// V_WMMA_F32_16X16X32_BF16 with fp32 accumulate.  Double-buffered LDS tiles
// (128x128 x BK=32) so next tile's global loads overlap the WMMA block.
// wTrans==0: W is [K,N] (ldw=N).  wTrans==1: W element (k,n) at W[n*ldw+k].
__global__ __launch_bounds__(256) void gemm_kernel(
    const float* __restrict__ A, const float* __restrict__ W,
    const float* __restrict__ bias, const float* __restrict__ residual,
    float* __restrict__ C, int M, int K, int N, int ldw, int wTrans, int act)
{
    __shared__ __align__(16) __bf16 ldsA[2][128 * 40];
    __shared__ __align__(16) __bf16 ldsB[2][128 * 40];

    const int tid  = threadIdx.x;
    const int lane = tid & 31;
    const int wid  = tid >> 5;
    const int hl   = lane >> 4;
    const int l15  = lane & 15;
    const int waveM = wid >> 1;
    const int waveN = wid & 1;
    const int m0 = blockIdx.y * 128;
    const int n0 = blockIdx.x * 128;

    float4 fa[4], fb[4];   // in-flight tile registers

    auto loadA = [&](int kb) {
        #pragma unroll
        for (int i = 0; i < 4; ++i) {
            int idx = tid + 256 * i;
            int row = idx >> 3;
            int c4  = idx & 7;
            fa[i] = *(const float4*)(A + (size_t)(m0 + row) * K + kb + c4 * 4);
        }
        if (kb + 32 < K)   // warm L2 one tile ahead
            __builtin_prefetch(A + (size_t)(m0 + (tid >> 3)) * K + kb + 32 + (tid & 7) * 4, 0, 3);
    };
    auto storeA = [&](int p) {
        #pragma unroll
        for (int i = 0; i < 4; ++i) {
            int idx = tid + 256 * i;
            int row = idx >> 3;
            int c4  = idx & 7;
            *(bf16x4*)(&ldsA[p][row * 40 + c4 * 4]) = tobf4(fa[i]);
        }
    };
    auto loadB = [&](int kb) {
        if (!wTrans) {
            #pragma unroll
            for (int i = 0; i < 4; ++i) {
                int idx = tid + 256 * i;
                int kk  = idx >> 5;
                int n4  = idx & 31;
                fb[i] = *(const float4*)(W + (size_t)(kb + kk) * ldw + n0 + n4 * 4);
            }
            if (kb + 32 < K)
                __builtin_prefetch(W + (size_t)(kb + 32 + (tid >> 5)) * ldw + n0 + (tid & 31) * 4, 0, 3);
        } else {
            #pragma unroll
            for (int i = 0; i < 4; ++i) {
                int idx = tid + 256 * i;
                int n   = idx >> 3;
                int c4  = idx & 7;
                fb[i] = *(const float4*)(W + (size_t)(n0 + n) * ldw + kb + c4 * 4);
            }
            if (kb + 32 < K)
                __builtin_prefetch(W + (size_t)(n0 + (tid >> 3)) * ldw + kb + 32 + (tid & 7) * 4, 0, 3);
        }
    };
    auto storeB = [&](int p) {
        if (!wTrans) {
            #pragma unroll
            for (int i = 0; i < 4; ++i) {
                int idx = tid + 256 * i;
                int kk  = idx >> 5;
                int n4  = idx & 31;
                ldsB[p][(n4 * 4 + 0) * 40 + kk] = tobf(fb[i].x);
                ldsB[p][(n4 * 4 + 1) * 40 + kk] = tobf(fb[i].y);
                ldsB[p][(n4 * 4 + 2) * 40 + kk] = tobf(fb[i].z);
                ldsB[p][(n4 * 4 + 3) * 40 + kk] = tobf(fb[i].w);
            }
        } else {
            #pragma unroll
            for (int i = 0; i < 4; ++i) {
                int idx = tid + 256 * i;
                int n   = idx >> 3;
                int c4  = idx & 7;
                *(bf16x4*)(&ldsB[p][n * 40 + c4 * 4]) = tobf4(fb[i]);
            }
        }
    };

    v8f zero = {0.f,0.f,0.f,0.f,0.f,0.f,0.f,0.f};
    v8f acc[2][4];
    #pragma unroll
    for (int mt = 0; mt < 2; ++mt)
        #pragma unroll
        for (int nt = 0; nt < 4; ++nt) acc[mt][nt] = zero;

    loadA(0);
    loadB(0);
    int p = 0;
    for (int k0 = 0; k0 < K; k0 += 32) {
        storeA(p);
        storeB(p);
        __syncthreads();
        if (k0 + 32 < K) { loadA(k0 + 32); loadB(k0 + 32); }   // overlap w/ WMMA

        ABFrag a[2], b[4];
        #pragma unroll
        for (int mt = 0; mt < 2; ++mt) {
            int r = waveM * 32 + mt * 16 + l15;
            a[mt].h[0] = *(const bf16x8*)(&ldsA[p][r * 40 + hl * 8]);
            a[mt].h[1] = *(const bf16x8*)(&ldsA[p][r * 40 + hl * 8 + 16]);
        }
        #pragma unroll
        for (int nt = 0; nt < 4; ++nt) {
            int n = waveN * 64 + nt * 16 + l15;
            b[nt].h[0] = *(const bf16x8*)(&ldsB[p][n * 40 + hl * 8]);
            b[nt].h[1] = *(const bf16x8*)(&ldsB[p][n * 40 + hl * 8 + 16]);
        }
        #pragma unroll
        for (int mt = 0; mt < 2; ++mt)
            #pragma unroll
            for (int nt = 0; nt < 4; ++nt)
                acc[mt][nt] = __builtin_amdgcn_wmma_f32_16x16x32_bf16(
                    false, a[mt].v, false, b[nt].v, (short)0, acc[mt][nt],
                    false, false);
        p ^= 1;
    }

    #pragma unroll
    for (int mt = 0; mt < 2; ++mt) {
        #pragma unroll
        for (int nt = 0; nt < 4; ++nt) {
            int colg = n0 + waveN * 64 + nt * 16 + l15;
            float bv = bias ? bias[colg] : 0.f;
            #pragma unroll
            for (int r = 0; r < 8; ++r) {
                int rowg = m0 + waveM * 32 + mt * 16 + r + hl * 8;
                float v = acc[mt][nt][r] + bv;
                if (act == 1) v = v / (1.f + __expf(-v));   // SiLU
                if (residual) v += residual[(size_t)rowg * N + colg];
                C[(size_t)rowg * N + colg] = v;
            }
        }
    }
}

// ---------------------------------------------------------------- attention
// Flash-style causal attention.  grid = (T/64, H), block = 128 (4 waves).
__global__ __launch_bounds__(128) void attn_kernel(
    const float* __restrict__ qkv, float* __restrict__ ao)
{
    __shared__ __align__(16) __bf16 ldsK[64 * 72];   // [kv][hd]
    __shared__ __align__(16) __bf16 ldsV[64 * 72];   // [hd][kv] (transposed)
    __shared__ __align__(16) __bf16 ldsP[64 * 72];   // per-wave P rows

    const int tid  = threadIdx.x;
    const int lane = tid & 31;
    const int w    = tid >> 5;
    const int hl   = lane >> 4;
    const int l15  = lane & 15;
    const int qb   = blockIdx.x * 64;
    const int hh   = blockIdx.y;
    const int ld   = 3 * D_;
    const int qcol = hh * HD_;

    ABFrag qf[2];
    {
        int qrow = qb + w * 16 + l15;
        const float* qp = qkv + (size_t)qrow * ld + qcol;
        #pragma unroll
        for (int ks = 0; ks < 2; ++ks) {
            qf[ks].h[0] = load8bf(qp + ks * 32 + hl * 8);
            qf[ks].h[1] = load8bf(qp + ks * 32 + hl * 8 + 16);
        }
    }

    v8f zero = {0.f,0.f,0.f,0.f,0.f,0.f,0.f,0.f};
    v8f o[4];
    #pragma unroll
    for (int ht = 0; ht < 4; ++ht) o[ht] = zero;
    float mrow[8], lrow[8];
    #pragma unroll
    for (int r = 0; r < 8; ++r) { mrow[r] = -INFINITY; lrow[r] = 0.f; }

    for (int kb = 0; kb <= qb; kb += 64) {
        // ---- stage K: batch-issue loads, then convert+store
        {
            float4 kreg[8];
            #pragma unroll
            for (int i = 0; i < 8; ++i) {
                int idx = tid + 128 * i;
                int kv  = idx >> 4;
                int c4  = idx & 15;
                kreg[i] = *(const float4*)(qkv + (size_t)(kb + kv) * ld + D_ + qcol + c4 * 4);
            }
            if (kb + 64 <= qb)
                __builtin_prefetch(qkv + (size_t)(kb + 64 + (tid >> 4)) * ld + D_ + qcol + (tid & 15) * 4, 0, 3);
            #pragma unroll
            for (int i = 0; i < 8; ++i) {
                int idx = tid + 128 * i;
                int kv  = idx >> 4;
                int c4  = idx & 15;
                *(bf16x4*)(ldsK + kv * 72 + c4 * 4) = tobf4(kreg[i]);
            }
        }
        // ---- stage V transposed
        {
            float4 vreg[8];
            #pragma unroll
            for (int i = 0; i < 8; ++i) {
                int idx = tid + 128 * i;
                int kv  = idx >> 4;
                int c4  = idx & 15;
                vreg[i] = *(const float4*)(qkv + (size_t)(kb + kv) * ld + 2 * D_ + qcol + c4 * 4);
            }
            if (kb + 64 <= qb)
                __builtin_prefetch(qkv + (size_t)(kb + 64 + (tid >> 4)) * ld + 2 * D_ + qcol + (tid & 15) * 4, 0, 3);
            #pragma unroll
            for (int i = 0; i < 8; ++i) {
                int idx = tid + 128 * i;
                int kv  = idx >> 4;
                int c4  = idx & 15;
                ldsV[(c4 * 4 + 0) * 72 + kv] = tobf(vreg[i].x);
                ldsV[(c4 * 4 + 1) * 72 + kv] = tobf(vreg[i].y);
                ldsV[(c4 * 4 + 2) * 72 + kv] = tobf(vreg[i].z);
                ldsV[(c4 * 4 + 3) * 72 + kv] = tobf(vreg[i].w);
            }
        }
        __syncthreads();

        // ---- S = Q * K^T
        v8f s[4];
        #pragma unroll
        for (int nt = 0; nt < 4; ++nt) s[nt] = zero;
        #pragma unroll
        for (int nt = 0; nt < 4; ++nt) {
            #pragma unroll
            for (int ks = 0; ks < 2; ++ks) {
                ABFrag bk;
                int n = nt * 16 + l15;
                bk.h[0] = *(const bf16x8*)(ldsK + n * 72 + ks * 32 + hl * 8);
                bk.h[1] = *(const bf16x8*)(ldsK + n * 72 + ks * 32 + hl * 8 + 16);
                s[nt] = __builtin_amdgcn_wmma_f32_16x16x32_bf16(
                    false, qf[ks].v, false, bk.v, (short)0, s[nt], false, false);
            }
        }

        // ---- scale + causal mask
        const float sc = 0.125f;   // 1/sqrt(64)
        if (kb == qb) {
            #pragma unroll
            for (int nt = 0; nt < 4; ++nt) {
                int col = kb + nt * 16 + l15;
                #pragma unroll
                for (int r = 0; r < 8; ++r) {
                    int row = qb + w * 16 + r + hl * 8;
                    s[nt][r] = (col <= row) ? s[nt][r] * sc : -INFINITY;
                }
            }
        } else {
            #pragma unroll
            for (int nt = 0; nt < 4; ++nt)
                #pragma unroll
                for (int r = 0; r < 8; ++r) s[nt][r] *= sc;
        }

        // ---- online softmax over 16-lane half groups
        float mx[8], mnew[8], rs[8];
        #pragma unroll
        for (int r = 0; r < 8; ++r)
            mx[r] = fmaxf(fmaxf(s[0][r], s[1][r]), fmaxf(s[2][r], s[3][r]));
        #pragma unroll
        for (int d = 1; d < 16; d <<= 1)
            #pragma unroll
            for (int r = 0; r < 8; ++r) mx[r] = fmaxf(mx[r], __shfl_xor(mx[r], d, 32));
        #pragma unroll
        for (int r = 0; r < 8; ++r) { mnew[r] = fmaxf(mrow[r], mx[r]); rs[r] = 0.f; }
        #pragma unroll
        for (int nt = 0; nt < 4; ++nt)
            #pragma unroll
            for (int r = 0; r < 8; ++r) {
                float e = __expf(s[nt][r] - mnew[r]);
                s[nt][r] = e; rs[r] += e;
            }
        #pragma unroll
        for (int d = 1; d < 16; d <<= 1)
            #pragma unroll
            for (int r = 0; r < 8; ++r) rs[r] += __shfl_xor(rs[r], d, 32);
        #pragma unroll
        for (int r = 0; r < 8; ++r) {
            float al = __expf(mrow[r] - mnew[r]);
            lrow[r] = lrow[r] * al + rs[r];
            mrow[r] = mnew[r];
            #pragma unroll
            for (int ht = 0; ht < 4; ++ht) o[ht][r] *= al;
        }

        // ---- write P (C-layout) to this wave's LDS region
        #pragma unroll
        for (int nt = 0; nt < 4; ++nt)
            #pragma unroll
            for (int r = 0; r < 8; ++r)
                ldsP[(w * 16 + r + hl * 8) * 72 + nt * 16 + l15] = tobf(s[nt][r]);
        asm volatile("s_wait_dscnt 0" ::: "memory");

        // ---- O += P * V
        #pragma unroll
        for (int ks = 0; ks < 2; ++ks) {
            ABFrag pf;
            int pr = w * 16 + l15;
            pf.h[0] = *(const bf16x8*)(ldsP + pr * 72 + ks * 32 + hl * 8);
            pf.h[1] = *(const bf16x8*)(ldsP + pr * 72 + ks * 32 + hl * 8 + 16);
            #pragma unroll
            for (int ht = 0; ht < 4; ++ht) {
                ABFrag vf;
                int n = ht * 16 + l15;
                vf.h[0] = *(const bf16x8*)(ldsV + n * 72 + ks * 32 + hl * 8);
                vf.h[1] = *(const bf16x8*)(ldsV + n * 72 + ks * 32 + hl * 8 + 16);
                o[ht] = __builtin_amdgcn_wmma_f32_16x16x32_bf16(
                    false, pf.v, false, vf.v, (short)0, o[ht], false, false);
            }
        }
        __syncthreads();
    }

    #pragma unroll
    for (int ht = 0; ht < 4; ++ht) {
        int col = hh * HD_ + ht * 16 + l15;
        #pragma unroll
        for (int r = 0; r < 8; ++r) {
            int row = qb + w * 16 + r + hl * 8;
            ao[(size_t)row * D_ + col] = o[ht][r] / lrow[r];
        }
    }
}

// ---------------------------------------------------------------- launch
extern "C" void kernel_launch(void* const* d_in, const int* in_sizes, int n_in,
                              void* d_out, int out_size, void* d_ws, size_t ws_size,
                              hipStream_t stream)
{
    const int*   ids   = (const int*)  d_in[0];
    const float* te    = (const float*)d_in[1];
    const float* pe    = (const float*)d_in[2];
    const float* qkv_w = (const float*)d_in[3];
    const float* qkv_b = (const float*)d_in[4];
    const float* out_w = (const float*)d_in[5];
    const float* out_b = (const float*)d_in[6];
    const float* ln1_w = (const float*)d_in[7];
    const float* ln1_b = (const float*)d_in[8];
    const float* ln2_w = (const float*)d_in[9];
    const float* ln2_b = (const float*)d_in[10];
    const float* up_w  = (const float*)d_in[11];
    const float* up_b  = (const float*)d_in[12];
    const float* dn_w  = (const float*)d_in[13];
    const float* dn_b  = (const float*)d_in[14];
    const float* lnf_w = (const float*)d_in[15];
    const float* lnf_b = (const float*)d_in[16];
    float* logits = (float*)d_out;

    float* ws = (float*)d_ws;
    float* x    = ws;                              // T*D
    float* h    = ws + (size_t)1 * T_ * D_;        // T*D
    float* qkvb = ws + (size_t)2 * T_ * D_;        // T*3D
    float* ao   = ws + (size_t)5 * T_ * D_;        // T*D
    float* up   = ws + (size_t)6 * T_ * D_;        // T*DF

    embed_kernel<<<T_, 256, 0, stream>>>(ids, te, pe, x);

    for (int i = 0; i < L_; ++i) {
        const float* qw = qkv_w + (size_t)i * D_ * 3 * D_;
        const float* ow = out_w + (size_t)i * D_ * D_;
        const float* uw = up_w  + (size_t)i * D_ * DF_;
        const float* dw = dn_w  + (size_t)i * DF_ * D_;

        ln_kernel<<<T_, 256, 0, stream>>>(x, ln1_w + i * D_, ln1_b + i * D_, h);
        gemm_kernel<<<dim3(3 * D_ / 128, T_ / 128), 256, 0, stream>>>(
            h, qw, qkv_b + (size_t)i * 3 * D_, nullptr, qkvb,
            T_, D_, 3 * D_, 3 * D_, 0, 0);
        attn_kernel<<<dim3(T_ / 64, H_), 128, 0, stream>>>(qkvb, ao);
        gemm_kernel<<<dim3(D_ / 128, T_ / 128), 256, 0, stream>>>(
            ao, ow, out_b + (size_t)i * D_, x, x,
            T_, D_, D_, D_, 0, 0);
        ln_kernel<<<T_, 256, 0, stream>>>(x, ln2_w + i * D_, ln2_b + i * D_, h);
        gemm_kernel<<<dim3(DF_ / 128, T_ / 128), 256, 0, stream>>>(
            h, uw, up_b + (size_t)i * DF_, nullptr, up,
            T_, D_, DF_, DF_, 0, 1);
        gemm_kernel<<<dim3(D_ / 128, T_ / 128), 256, 0, stream>>>(
            up, dw, dn_b + (size_t)i * D_, x, x,
            T_, DF_, D_, D_, 0, 0);
    }

    ln_kernel<<<T_, 256, 0, stream>>>(x, lnf_w, lnf_b, h);
    gemm_kernel<<<dim3(V_ / 128, T_ / 128), 256, 0, stream>>>(
        h, te, nullptr, nullptr, logits,
        T_, D_, V_, D_, 1, 0);
}
